// SurnameClassifierRNN_42417097016405
// MI455X (gfx1250) — matrix operand order repaired and checked by reading, not compiled, not run on hardware
//
#include <hip/hip_runtime.h>
#include <hip/hip_bf16.h>
#include <math.h>

#define BN 4096
#define SN 64
#define EN 128
#define HN 256
#define CN 18

typedef _Float16 v16h __attribute__((ext_vector_type(16)));
typedef float    v8f  __attribute__((ext_vector_type(8)));

union HFrag { v16h v; _Float16 h[16]; };
union FAcc  { v8f  v; float    f[8];  };

__device__ __forceinline__ _Float16 f2h(float x) { return (_Float16)x; }
__device__ __forceinline__ float    h2f(_Float16 x) { return (float)x; }

// ---------------------------------------------------------------------------
// Phase 0: convert weights fp32 -> f16 in workspace, zero initial hidden
// ---------------------------------------------------------------------------
__global__ void __launch_bounds__(256)
prep_kernel(const float* __restrict__ Wih, const float* __restrict__ Whh,
            const float* __restrict__ W1,
            _Float16* __restrict__ WihH, _Float16* __restrict__ WhhH,
            _Float16* __restrict__ W1H, _Float16* __restrict__ h0) {
    int i = blockIdx.x * blockDim.x + threadIdx.x;
    if (i < HN * EN) WihH[i] = f2h(Wih[i]);
    int j = i - HN * EN;
    if (j >= 0 && j < HN * HN) WhhH[j] = f2h(Whh[j]);
    int k = j - HN * HN;
    if (k >= 0 && k < HN * HN) W1H[k] = f2h(W1[k]);
    int m = k - HN * HN;
    if (m >= 0 && m < BN * HN) h0[m] = (_Float16)0.0f;
}

// ---------------------------------------------------------------------------
// Phase 1: embedding gather + input projection
//   xW[s*BN + b, h] = sum_e emb[x_in[b,s], e] * W_ih[h, e] + b_ih[h]
// Rows ordered r = s*BN + b so stores into xW ([S,B,H] layout) are contiguous.
// One block = 16 rows x 256 cols. 8 waves, 2 N-tiles each, K-loop over E=128.
// ---------------------------------------------------------------------------
__global__ void __launch_bounds__(256)
embed_proj_kernel(const int* __restrict__ x_in, const float* __restrict__ emb,
                  const _Float16* __restrict__ WihH, const float* __restrict__ b_ih,
                  _Float16* __restrict__ xW) {
    __shared__ __align__(16) _Float16 Abuf[16 * EN];  // 4 KB
    __shared__ int tok[16];
    const int tid  = threadIdx.x;
    const int row0 = blockIdx.x * 16;   // r = s*BN + b
    const int s    = row0 / BN;
    const int b0   = row0 % BN;
    if (tid < 16) tok[tid] = x_in[(b0 + tid) * SN + s];
    __syncthreads();
    // 16 rows x 128 floats -> f16 LDS, coalesced
    #pragma unroll
    for (int i = 0; i < 8; ++i) {
        int e = i * 256 + tid;
        int m = e >> 7, k = e & 127;
        Abuf[e] = f2h(emb[(size_t)tok[m] * EN + k]);
    }
    __syncthreads();

    const int lane = tid & 31, wave = tid >> 5;
    const int M  = lane & 15;   // A row / C row-sub
    const int hi = lane >> 4;
    const int Nc = lane & 15;   // C/D column

    for (int nt = wave * 2; nt < wave * 2 + 2; ++nt) {
        const int n = nt * 16 + Nc;
        FAcc acc;
        #pragma unroll
        for (int r = 0; r < 8; ++r) acc.f[r] = 0.0f;

        #pragma unroll
        for (int kc = 0; kc < EN / 32; ++kc) {
            HFrag a, bb;
            #pragma unroll
            for (int j = 0; j < 8; ++j) {
                a.h[j]     = Abuf[M * EN + kc * 32 + hi * 8 + j];
                a.h[j + 8] = Abuf[M * EN + kc * 32 + 16 + hi * 8 + j];
            }
            #pragma unroll
            for (int j = 0; j < 8; ++j) {
                bb.h[j]     = WihH[(size_t)n * EN + kc * 32 + hi * 8 + j];
                bb.h[j + 8] = WihH[(size_t)n * EN + kc * 32 + 16 + hi * 8 + j];
            }
            acc.v = __builtin_amdgcn_wmma_f32_16x16x32_f16(
                false, a.v, false, bb.v, (short)0, acc.v, false, false);
        }
        const float bias = b_ih[n];
        #pragma unroll
        for (int r = 0; r < 8; ++r) {
            int row = row0 + r + hi * 8;
            xW[(size_t)row * HN + n] = f2h(acc.f[r] + bias);
        }
    }
}

// ---------------------------------------------------------------------------
// Phase 2 (x64, sequential): h_t = tanh(xW[t] + h_{t-1} @ W_hh^T + b_hh)
// Also scatters h into `last` for rows where t == len-1.
// ---------------------------------------------------------------------------
__global__ void __launch_bounds__(256)
rnn_step_kernel(int t, const _Float16* __restrict__ hprev,
                _Float16* __restrict__ hnext, const _Float16* __restrict__ WhhH,
                const _Float16* __restrict__ xW, const float* __restrict__ b_hh,
                const int* __restrict__ x_len, float* __restrict__ last) {
    __shared__ __align__(16) _Float16 Abuf[16 * HN];  // 8 KB
    __shared__ int lens[16];
    const int tid  = threadIdx.x;
    const int row0 = blockIdx.x * 16;
    if (tid < 16) lens[tid] = x_len[row0 + tid];
    // 16 rows x 256 f16 = 8 KB, vectorized copy
    const uint2* src = (const uint2*)(hprev + (size_t)row0 * HN);
    uint2* dst = (uint2*)Abuf;
    #pragma unroll
    for (int i = 0; i < 4; ++i) dst[i * 256 + tid] = src[i * 256 + tid];
    __syncthreads();

    const int lane = tid & 31, wave = tid >> 5;
    const int M  = lane & 15;
    const int hi = lane >> 4;
    const int Nc = lane & 15;

    for (int nt = wave * 2; nt < wave * 2 + 2; ++nt) {
        const int n = nt * 16 + Nc;
        FAcc acc;
        #pragma unroll
        for (int r = 0; r < 8; ++r) acc.f[r] = 0.0f;

        #pragma unroll
        for (int kc = 0; kc < HN / 32; ++kc) {
            HFrag a, bb;
            #pragma unroll
            for (int j = 0; j < 8; ++j) {
                a.h[j]     = Abuf[M * HN + kc * 32 + hi * 8 + j];
                a.h[j + 8] = Abuf[M * HN + kc * 32 + 16 + hi * 8 + j];
            }
            #pragma unroll
            for (int j = 0; j < 8; ++j) {
                bb.h[j]     = WhhH[(size_t)n * HN + kc * 32 + hi * 8 + j];
                bb.h[j + 8] = WhhH[(size_t)n * HN + kc * 32 + 16 + hi * 8 + j];
            }
            acc.v = __builtin_amdgcn_wmma_f32_16x16x32_f16(
                false, a.v, false, bb.v, (short)0, acc.v, false, false);
        }
        const float bias = b_hh[n];
        #pragma unroll
        for (int r = 0; r < 8; ++r) {
            int row = row0 + r + hi * 8;
            float v = acc.f[r] + bias +
                      h2f(xW[(size_t)t * BN * HN + (size_t)row * HN + n]);
            float h = tanhf(v);
            hnext[(size_t)row * HN + n] = f2h(h);
            if (lens[r + hi * 8] - 1 == t) last[(size_t)row * HN + n] = h;
        }
    }
}

// ---------------------------------------------------------------------------
// Phase 3: y = relu(last @ W1^T + b1);  out = y @ W2^T + b2
// WMMA for the HxH layer; tiny 18-wide layer via VALU dots from LDS.
// ---------------------------------------------------------------------------
__global__ void __launch_bounds__(256)
classifier_kernel(const float* __restrict__ last, const _Float16* __restrict__ W1H,
                  const float* __restrict__ b1, const float* __restrict__ W2,
                  const float* __restrict__ b2, float* __restrict__ out) {
    __shared__ __align__(16) _Float16 Abuf[16 * HN];  // 8 KB
    __shared__ float Ybuf[16 * HN];                   // 16 KB
    const int tid  = threadIdx.x;
    const int row0 = blockIdx.x * 16;
    #pragma unroll
    for (int i = 0; i < 16; ++i) {
        int e = i * 256 + tid;
        Abuf[e] = f2h(last[(size_t)row0 * HN + e]);
    }
    __syncthreads();

    const int lane = tid & 31, wave = tid >> 5;
    const int M  = lane & 15;
    const int hi = lane >> 4;
    const int Nc = lane & 15;

    for (int nt = wave * 2; nt < wave * 2 + 2; ++nt) {
        const int n = nt * 16 + Nc;
        FAcc acc;
        #pragma unroll
        for (int r = 0; r < 8; ++r) acc.f[r] = 0.0f;

        #pragma unroll
        for (int kc = 0; kc < HN / 32; ++kc) {
            HFrag a, bb;
            #pragma unroll
            for (int j = 0; j < 8; ++j) {
                a.h[j]     = Abuf[M * HN + kc * 32 + hi * 8 + j];
                a.h[j + 8] = Abuf[M * HN + kc * 32 + 16 + hi * 8 + j];
            }
            #pragma unroll
            for (int j = 0; j < 8; ++j) {
                bb.h[j]     = W1H[(size_t)n * HN + kc * 32 + hi * 8 + j];
                bb.h[j + 8] = W1H[(size_t)n * HN + kc * 32 + 16 + hi * 8 + j];
            }
            acc.v = __builtin_amdgcn_wmma_f32_16x16x32_f16(
                false, a.v, false, bb.v, (short)0, acc.v, false, false);
        }
        const float bias = b1[n];
        #pragma unroll
        for (int r = 0; r < 8; ++r) {
            float y = acc.f[r] + bias;
            Ybuf[(r + hi * 8) * HN + n] = y > 0.0f ? y : 0.0f;
        }
    }
    __syncthreads();

    for (int task = tid; task < 16 * CN; task += 256) {
        int m = task / CN, c = task % CN;
        float sum = b2[c];
        #pragma unroll 4
        for (int k = 0; k < HN; ++k) sum += Ybuf[m * HN + k] * W2[c * HN + k];
        out[(size_t)(row0 + m) * CN + c] = sum;
    }
}

// ---------------------------------------------------------------------------
extern "C" void kernel_launch(void* const* d_in, const int* in_sizes, int n_in,
                              void* d_out, int out_size, void* d_ws, size_t ws_size,
                              hipStream_t stream) {
    (void)in_sizes; (void)n_in; (void)out_size; (void)ws_size;
    const int*   x_in  = (const int*)  d_in[0];
    const int*   x_len = (const int*)  d_in[1];
    const float* emb   = (const float*)d_in[2];
    const float* W_ih  = (const float*)d_in[3];
    const float* W_hh  = (const float*)d_in[4];
    const float* b_ih  = (const float*)d_in[5];
    const float* b_hh  = (const float*)d_in[6];
    const float* W1    = (const float*)d_in[7];
    const float* b1    = (const float*)d_in[8];
    const float* W2    = (const float*)d_in[9];
    const float* b2    = (const float*)d_in[10];
    float* out = (float*)d_out;

    char* ws = (char*)d_ws;
    _Float16* WihH = (_Float16*)(ws + 0);            //  64 KB
    _Float16* WhhH = (_Float16*)(ws + 65536);        // 128 KB
    _Float16* W1H  = (_Float16*)(ws + 196608);       // 128 KB
    _Float16* h0   = (_Float16*)(ws + 327680);       //   2 MB
    _Float16* h1   = (_Float16*)(ws + 2425088);      //   2 MB
    float*    last = (float*)   (ws + 4522240);      //   4 MB
    _Float16* xW   = (_Float16*)(ws + 8716544);      // 128 MB  [S,B,H] f16

    const int prepN = HN * EN + HN * HN + HN * HN + BN * HN;
    prep_kernel<<<(prepN + 255) / 256, 256, 0, stream>>>(
        W_ih, W_hh, W1, WihH, WhhH, W1H, h0);

    embed_proj_kernel<<<SN * BN / 16, 256, 0, stream>>>(x_in, emb, WihH, b_ih, xW);

    _Float16* hp = h0;
    _Float16* hn = h1;
    for (int t = 0; t < SN; ++t) {
        rnn_step_kernel<<<BN / 16, 256, 0, stream>>>(
            t, hp, hn, WhhH, xW, b_hh, x_len, last);
        _Float16* tmp = hp; hp = hn; hn = tmp;
    }

    classifier_kernel<<<BN / 16, 256, 0, stream>>>(last, W1H, b1, W2, b2, out);
}
